// CausalSelfAttention_43705587204376
// MI455X (gfx1250) — matrix operand order repaired and checked by reading, not compiled, not run on hardware
//
#include <hip/hip_runtime.h>
#include <hip/hip_bf16.h>
#include <stdint.h>

// ---------------------------------------------------------------------------
// CausalSelfAttention forward for MI455X (gfx1250, wave32, WMMA)
//   B=16, T=1024, C=768, H=12, D=64
//   qkv = x@W_qkv + b ; causal SDPA ; y = att@W_proj + b
// Strategy: bf16 data path with f32 WMMA accumulation (v_wmma_f32_16x16x32_bf16)
// ---------------------------------------------------------------------------

typedef __bf16 bf16;
typedef __attribute__((ext_vector_type(16))) __bf16 v16bf;
typedef __attribute__((ext_vector_type(8)))  float  v8f;

union FragU { v16bf v; uint4 u2[2]; };

#define HDI __device__ __forceinline__

static HDI unsigned pack_bf16(float a, float b) {
  union { __bf16 h; unsigned short s; } ua, ub;
  ua.h = (__bf16)a; ub.h = (__bf16)b;
  return (unsigned)ua.s | ((unsigned)ub.s << 16);
}

static HDI v8f wmma_bf16(const FragU& a, const FragU& b, v8f c) {
  // D = A(16x32 bf16) * B(32x16 bf16) + C(16x16 f32)
  return __builtin_amdgcn_wmma_f32_16x16x32_bf16(
      /*neg_a=*/false, a.v, /*neg_b=*/false, b.v,
      /*c_mod=*/(short)0, c, /*reuse_a=*/false, /*reuse_b=*/false);
}

// ---------------------------------------------------------------------------
// Pack kernels
// ---------------------------------------------------------------------------
__global__ void cvt_f32_bf16(const float* __restrict__ src,
                             bf16* __restrict__ dst, int n4) {
  int i = blockIdx.x * blockDim.x + threadIdx.x;
  if (i < n4) {
    float4 f = ((const float4*)src)[i];
    unsigned lo = pack_bf16(f.x, f.y);
    unsigned hi = pack_bf16(f.z, f.w);
    ((uint2*)dst)[i] = make_uint2(lo, hi);
  }
}

// dst[j*K + k] = (bf16) src[k*N + j]   (W stored [K,N] row-major -> Wt [N,K])
__global__ void tr_cvt(const float* __restrict__ src, bf16* __restrict__ dst,
                       int K, int N) {
  long idx = (long)blockIdx.x * blockDim.x + threadIdx.x;
  long total = (long)K * N;
  if (idx < total) {
    int j = (int)(idx / K);
    int k = (int)(idx - (long)j * K);
    dst[idx] = (bf16)src[(long)k * N + j];
  }
}

// ---------------------------------------------------------------------------
// QKV GEMM: [16384 x 768] bf16 @ Wt[2304 x 768]^T -> Q/K [B,H,T,D], Vt [B,H,D,T]
// Per wave: 64x64 macrotile (4x4 WMMA tiles). Block = 4 waves -> 128x128.
// ---------------------------------------------------------------------------
__global__ __launch_bounds__(128) void qkv_gemm(
    const bf16* __restrict__ A, const bf16* __restrict__ Bt,
    const float* __restrict__ bias,
    bf16* __restrict__ Qb, bf16* __restrict__ Kb, bf16* __restrict__ Vt) {
  const int Kd = 768;
  int lane = threadIdx.x & 31;
  int w    = threadIdx.x >> 5;
  int ml   = lane & 15;
  int hi   = lane >> 4;
  int m0 = blockIdx.y * 128 + (w >> 1) * 64;
  int n0 = blockIdx.x * 128 + (w & 1) * 64;

  const v8f z8 = {0.f,0.f,0.f,0.f,0.f,0.f,0.f,0.f};
  v8f acc[4][4];
#pragma unroll
  for (int mt = 0; mt < 4; mt++)
#pragma unroll
    for (int nt = 0; nt < 4; nt++) acc[mt][nt] = z8;

  for (int k0 = 0; k0 < Kd; k0 += 32) {
    FragU a[4];
#pragma unroll
    for (int mt = 0; mt < 4; mt++) {
      // A layout 16x32: e0..7 -> k = k0+hi*8+e ; e8..15 -> k = k0+hi*8+16+e'
      const bf16* p = A + (size_t)(m0 + mt * 16 + ml) * Kd + k0 + hi * 8;
      a[mt].u2[0] = *(const uint4*)p;
      a[mt].u2[1] = *(const uint4*)(p + 16);
    }
#pragma unroll
    for (int nt = 0; nt < 4; nt++) {
      FragU b;
      // B layout 32x16: lane = n, e0..7 -> k = k0+hi*8+e ; e8..15 -> k = k0+hi*8+16+e'
      const bf16* p = Bt + (size_t)(n0 + nt * 16 + ml) * Kd + k0 + hi * 8;
      b.u2[0] = *(const uint4*)p;
      b.u2[1] = *(const uint4*)(p + 16);
#pragma unroll
      for (int mt = 0; mt < 4; mt++)
        acc[mt][nt] = wmma_bf16(a[mt], b, acc[mt][nt]);
    }
  }

  // Epilogue: D layout -> lane = out col, VGPR r -> out row (r + hi*8)
#pragma unroll
  for (int nt = 0; nt < 4; nt++) {
    int j0   = n0 + nt * 16;        // wave-uniform
    int kind = j0 / 768;            // 0=Q 1=K 2=V
    int cw   = j0 - kind * 768;
    int h    = cw >> 6;
    int d    = (cw & 63) + ml;
    float bj = bias[j0 + ml];
#pragma unroll
    for (int mt = 0; mt < 4; mt++) {
      int gm0  = m0 + mt * 16;      // global row (token) base; tile never crosses batch
      int bidx = gm0 >> 10;
      int tl   = (gm0 & 1023) + hi * 8;
      if (kind == 2) {
        // V transposed store: Vt[(bh*64 + d)*1024 + t], t contiguous over r
        unsigned o0 = pack_bf16(acc[mt][nt][0] + bj, acc[mt][nt][1] + bj);
        unsigned o1 = pack_bf16(acc[mt][nt][2] + bj, acc[mt][nt][3] + bj);
        unsigned o2 = pack_bf16(acc[mt][nt][4] + bj, acc[mt][nt][5] + bj);
        unsigned o3 = pack_bf16(acc[mt][nt][6] + bj, acc[mt][nt][7] + bj);
        bf16* dst = Vt + ((size_t)(bidx * 12 + h) * 64 + d) * 1024 + tl;
        *(uint4*)dst = make_uint4(o0, o1, o2, o3);
      } else {
        bf16* dst = (kind == 0 ? Qb : Kb) +
                    ((size_t)(bidx * 12 + h) * 1024 + tl) * 64 + d;
#pragma unroll
        for (int r = 0; r < 8; r++)
          dst[(size_t)r * 64] = (bf16)(acc[mt][nt][r] + bj);
      }
    }
  }
}

// ---------------------------------------------------------------------------
// Flash attention (transposed): one wave per (head, 16-query tile).
//   S^T = K * Q^T   (D-layout: lane = query, VGPRs = keys)  -> per-lane softmax
//   O^T = V^T * P^T (accumulated, online-rescaled)
// ---------------------------------------------------------------------------
__global__ __launch_bounds__(32) void attn_kernel(
    const bf16* __restrict__ Q, const bf16* __restrict__ K,
    const bf16* __restrict__ Vt, bf16* __restrict__ att) {
  int lane = threadIdx.x & 31;
  int ml = lane & 15;
  int hi = lane >> 4;
  int bh = blockIdx.y;            // b*12 + h
  int t0 = blockIdx.x * 16;       // query tile base

  const bf16* Qp = Q  + (size_t)bh * 1024 * 64;
  const bf16* Kp = K  + (size_t)bh * 1024 * 64;
  const bf16* Vp = Vt + (size_t)bh * 64 * 1024;

  // Q^T B-fragments (2 k-steps over D=64): lane = query, d = hi*8.. and hi*8+16..
  FragU bq[2];
#pragma unroll
  for (int s = 0; s < 2; s++) {
    const bf16* p = Qp + (size_t)(t0 + ml) * 64 + s * 32 + hi * 8;
    bq[s].u2[0] = *(const uint4*)p;
    bq[s].u2[1] = *(const uint4*)(p + 16);
  }

  const v8f z8 = {0.f,0.f,0.f,0.f,0.f,0.f,0.f,0.f};
  v8f acc[4]; // O^T, 4 d-tiles of 16
#pragma unroll
  for (int md = 0; md < 4; md++) acc[md] = z8;
  float m_run = -1e30f, l_run = 0.f;

  for (int kc = 0; kc < t0 + 16; kc += 32) {
    // ---- S^T = K * Q^T over this 32-key chunk (two 16-key tiles) ----
    v8f st[2];
#pragma unroll
    for (int j = 0; j < 2; j++) {
      const bf16* p = Kp + (size_t)(kc + j * 16 + ml) * 64 + hi * 8;
      FragU ak0, ak1;
      ak0.u2[0] = *(const uint4*)p;         // d = hi*8 .. +7
      ak0.u2[1] = *(const uint4*)(p + 16);  // d = hi*8+16 ..
      ak1.u2[0] = *(const uint4*)(p + 32);
      ak1.u2[1] = *(const uint4*)(p + 48);
      v8f z = z8;
      z = wmma_bf16(ak0, bq[0], z);
      st[j] = wmma_bf16(ak1, bq[1], z);
    }

    // ---- scale + causal mask + online softmax (lane = query) ----
    bool need_mask = (kc + 31 > t0);  // wave-uniform
    float sc[2][8];
    float cmax = -1e30f;
#pragma unroll
    for (int j = 0; j < 2; j++)
#pragma unroll
      for (int r = 0; r < 8; r++) {
        float s = st[j][r] * 0.125f;  // 1/sqrt(64)
        if (need_mask) {
          int kabs = kc + j * 16 + r + hi * 8;
          int qabs = t0 + ml;
          if (kabs > qabs) s = -1e30f;
        }
        sc[j][r] = s;
        cmax = fmaxf(cmax, s);
      }
    cmax = fmaxf(cmax, __shfl_xor(cmax, 16, 32));
    float m_new = fmaxf(m_run, cmax);
    float alpha = __expf(m_run - m_new);

    float psum = 0.f;
    unsigned pk[2][4];
#pragma unroll
    for (int j = 0; j < 2; j++)
#pragma unroll
      for (int i = 0; i < 4; i++) {
        float p0 = __expf(sc[j][2 * i]     - m_new);
        float p1 = __expf(sc[j][2 * i + 1] - m_new);
        psum += p0 + p1;
        pk[j][i] = pack_bf16(p0, p1);
      }
    psum += __shfl_xor(psum, 16, 32);
    l_run = l_run * alpha + psum;
    m_run = m_new;
#pragma unroll
    for (int md = 0; md < 4; md++) acc[md] = acc[md] * alpha;

    // ---- build P^T B-fragment (32 keys x 16 queries): half-exchange ----
    FragU pf;
    unsigned* pd = (unsigned*)&pf;
#pragma unroll
    for (int i = 0; i < 4; i++) {
      unsigned x0 = (unsigned)__shfl_xor((int)pk[0][i], 16, 32);
      unsigned x1 = (unsigned)__shfl_xor((int)pk[1][i], 16, 32);
      pd[i]     = hi ? x1 : pk[0][i];  // e0..7 : keys hi?16..23:0..7
      pd[4 + i] = hi ? pk[1][i] : x0;  // e8..15: keys hi?24..31:8..15
    }

    // ---- O^T += V^T * P^T (4 d-tiles) ----
#pragma unroll
    for (int md = 0; md < 4; md++) {
      const bf16* p = Vp + (size_t)(md * 16 + ml) * 1024 + kc + hi * 16;
      FragU av;
      av.u2[0] = *(const uint4*)p;         // keys kc+hi*16 .. +7
      av.u2[1] = *(const uint4*)(p + 8);   // keys kc+hi*16+8 ..
      acc[md] = wmma_bf16(av, pf, acc[md]);
    }
  }

  // ---- normalize and store O^T: lane = query, packed b128 per d-tile ----
  float inv = 1.f / l_run;
  int b = bh / 12, h = bh - (bh / 12) * 12;
#pragma unroll
  for (int md = 0; md < 4; md++) {
    unsigned o0 = pack_bf16(acc[md][0] * inv, acc[md][1] * inv);
    unsigned o1 = pack_bf16(acc[md][2] * inv, acc[md][3] * inv);
    unsigned o2 = pack_bf16(acc[md][4] * inv, acc[md][5] * inv);
    unsigned o3 = pack_bf16(acc[md][6] * inv, acc[md][7] * inv);
    bf16* dst = att + (size_t)(b * 1024 + t0 + ml) * 768 + h * 64 +
                md * 16 + hi * 8;
    *(uint4*)dst = make_uint4(o0, o1, o2, o3);
  }
}

// ---------------------------------------------------------------------------
// Output projection: att[16384x768] bf16 @ WprojT[768x768]^T + bias -> f32 out
// ---------------------------------------------------------------------------
__global__ __launch_bounds__(128) void proj_gemm(
    const bf16* __restrict__ A, const bf16* __restrict__ Bt,
    const float* __restrict__ bias, float* __restrict__ Y) {
  const int Kd = 768, Nn = 768;
  int lane = threadIdx.x & 31;
  int w    = threadIdx.x >> 5;
  int ml   = lane & 15;
  int hi   = lane >> 4;
  int m0 = blockIdx.y * 128 + (w >> 1) * 64;
  int n0 = blockIdx.x * 128 + (w & 1) * 64;

  const v8f z8 = {0.f,0.f,0.f,0.f,0.f,0.f,0.f,0.f};
  v8f acc[4][4];
#pragma unroll
  for (int mt = 0; mt < 4; mt++)
#pragma unroll
    for (int nt = 0; nt < 4; nt++) acc[mt][nt] = z8;

  for (int k0 = 0; k0 < Kd; k0 += 32) {
    FragU a[4];
#pragma unroll
    for (int mt = 0; mt < 4; mt++) {
      const bf16* p = A + (size_t)(m0 + mt * 16 + ml) * Kd + k0 + hi * 8;
      a[mt].u2[0] = *(const uint4*)p;
      a[mt].u2[1] = *(const uint4*)(p + 16);
    }
#pragma unroll
    for (int nt = 0; nt < 4; nt++) {
      FragU b;
      const bf16* p = Bt + (size_t)(n0 + nt * 16 + ml) * Kd + k0 + hi * 8;
      b.u2[0] = *(const uint4*)p;
      b.u2[1] = *(const uint4*)(p + 16);
#pragma unroll
      for (int mt = 0; mt < 4; mt++)
        acc[mt][nt] = wmma_bf16(a[mt], b, acc[mt][nt]);
    }
  }

#pragma unroll
  for (int nt = 0; nt < 4; nt++) {
    int j = n0 + nt * 16 + ml;
    float bj = bias[j];
#pragma unroll
    for (int mt = 0; mt < 4; mt++) {
      int gm0 = m0 + mt * 16 + hi * 8;
#pragma unroll
      for (int r = 0; r < 8; r++)
        Y[(size_t)(gm0 + r) * Nn + j] = acc[mt][nt][r] + bj;
    }
  }
}

// ---------------------------------------------------------------------------
// Host launcher
// ---------------------------------------------------------------------------
extern "C" void kernel_launch(void* const* d_in, const int* in_sizes, int n_in,
                              void* d_out, int out_size, void* d_ws,
                              size_t ws_size, hipStream_t stream) {
  const float* x      = (const float*)d_in[0];   // [16,1024,768]
  const float* W_qkv  = (const float*)d_in[1];   // [768,2304]
  const float* b_qkv  = (const float*)d_in[2];   // [2304]
  const float* W_proj = (const float*)d_in[3];   // [768,768]
  const float* b_proj = (const float*)d_in[4];   // [768]
  float* out = (float*)d_out;                    // [16,1024,768] f32

  const int Bb = 16, T = 1024, C = 768, H = 12, D = 64;
  const int M = Bb * T;          // 16384
  const int N3 = 3 * C;          // 2304
  (void)in_sizes; (void)n_in; (void)out_size; (void)ws_size;

  // Workspace layout (bf16 buffers), all offsets 256B aligned
  char* ws = (char*)d_ws;
  size_t off = 0;
  bf16* xb     = (bf16*)(ws + off); off += (size_t)M * C * 2;       // 25.2 MB
  bf16* wqkvT  = (bf16*)(ws + off); off += (size_t)N3 * C * 2;      //  3.5 MB
  bf16* wprojT = (bf16*)(ws + off); off += (size_t)C * C * 2;       //  1.2 MB
  bf16* Qb     = (bf16*)(ws + off); off += (size_t)Bb * H * T * D * 2;
  bf16* Kb     = (bf16*)(ws + off); off += (size_t)Bb * H * T * D * 2;
  bf16* Vt     = (bf16*)(ws + off); off += (size_t)Bb * H * D * T * 2;
  bf16* att    = (bf16*)(ws + off); off += (size_t)M * C * 2;

  // 1) pack inputs to bf16 (weights transposed to [N,K])
  {
    int n4 = (M * C) / 4;
    cvt_f32_bf16<<<(n4 + 255) / 256, 256, 0, stream>>>(x, xb, n4);
    long tq = (long)C * N3;
    tr_cvt<<<(unsigned)((tq + 255) / 256), 256, 0, stream>>>(W_qkv, wqkvT, C, N3);
    long tp = (long)C * C;
    tr_cvt<<<(unsigned)((tp + 255) / 256), 256, 0, stream>>>(W_proj, wprojT, C, C);
  }

  // 2) fused QKV projection -> Q,K [B,H,T,D], V transposed [B,H,D,T]
  qkv_gemm<<<dim3(N3 / 128, M / 128), 128, 0, stream>>>(xb, wqkvT, b_qkv,
                                                        Qb, Kb, Vt);

  // 3) causal flash attention: one wave per (head, 16-query tile)
  attn_kernel<<<dim3(T / 16, Bb * H), 32, 0, stream>>>(Qb, Kb, Vt, att);

  // 4) output projection -> f32 d_out
  proj_gemm<<<dim3(C / 128, M / 128), 128, 0, stream>>>(att, wprojT, b_proj,
                                                        out);
}
